// PhaseLinear_28802050687546
// MI455X (gfx1250) — matrix operand order repaired and checked by reading, not compiled, not run on hardware
//
#include <hip/hip_runtime.h>
#include <hip/hip_bf16.h>

// ---------------------------------------------------------------------------
// PhaseLinear on MI455X (gfx1250):
//   out[b,n] = sum_p alpha[b,p] * ( x[b,:] . W_p[n,:] + bias_p[n] )
// Compute-bound (~860 FLOP/byte). Path:
//   1) prologue kernel: one-time fp32 -> bf16 hi/lo split of x and W into d_ws
//   2) GEMM kernel: double-buffered LDS staged with global_load_async_to_lds_b128
//      (ASYNCcnt), bf16x2 error-compensated WMMA (hi*hi + lo*hi + hi*lo),
//      per-expert spline blend folded into the accumulator chain.
// Fallback (ws too small): self-contained kernel converting in-loop.
// ---------------------------------------------------------------------------

typedef __bf16 bf16_t;
typedef bf16_t v16bf __attribute__((ext_vector_type(16)));
typedef float  v8f   __attribute__((ext_vector_type(8)));
typedef unsigned int u32;

#define IN_DIM   1024
#define OUT_DIM  1024
#define NUM_CP   4
#define KTILE    32      // K per WMMA (bf16)
#define WG_M     64      // block tile rows
#define WG_N     128     // block tile cols
#define AS       20      // LDS row stride in u32 (40 bf16 = 32 data + 8 pad; 80B, 16B-aligned)
#define NSTEPS   (NUM_CP * (IN_DIM / KTILE))   // 128 flattened (p, ktile) steps

union FragU { uint4 u[2]; v16bf v; };

__device__ __forceinline__ u32 bf16_rne(float f) {
    u32 u = __float_as_uint(f);
    return (u + 0x7FFFu + ((u >> 16) & 1u)) >> 16;
}
__device__ __forceinline__ float bf16_val(u32 h) { return __uint_as_float(h << 16); }

// split 8 fp32 -> 8 bf16 hi (packed uint4) + 8 bf16 lo (packed uint4)
__device__ __forceinline__ void split8(const float* v, uint4& H, uint4& L) {
    u32 h[8], l[8];
#pragma unroll
    for (int i = 0; i < 8; ++i) {
        h[i] = bf16_rne(v[i]);
        l[i] = bf16_rne(v[i] - bf16_val(h[i]));
    }
    H = make_uint4(h[0] | (h[1] << 16), h[2] | (h[3] << 16),
                   h[4] | (h[5] << 16), h[6] | (h[7] << 16));
    L = make_uint4(l[0] | (l[1] << 16), l[2] | (l[3] << 16),
                   l[4] | (l[5] << 16), l[6] | (l[7] << 16));
}

// async 16B copy: global -> LDS, tracked by ASYNCcnt (gfx1250)
__device__ __forceinline__ void async_b128(u32 lds_off, const bf16_t* g) {
    asm volatile("global_load_async_to_lds_b128 %0, %1, off"
                 :: "v"(lds_off), "v"(g) : "memory");
}
__device__ __forceinline__ void wait_async0() {
    asm volatile("s_wait_asynccnt 0x0" ::: "memory");
}

// ---------------------------------------------------------------------------
// Prologue: fp32 -> bf16 hi/lo, 8 elements per thread
// ---------------------------------------------------------------------------
__global__ __launch_bounds__(256)
void split_to_bf16(const float* __restrict__ src,
                   uint4* __restrict__ hi, uint4* __restrict__ lo, int n8) {
    const int i = blockIdx.x * blockDim.x + threadIdx.x;
    if (i >= n8) return;
    const float4 f0 = *(const float4*)(src + (size_t)i * 8);
    const float4 f1 = *(const float4*)(src + (size_t)i * 8 + 4);
    float v[8] = {f0.x, f0.y, f0.z, f0.w, f1.x, f1.y, f1.z, f1.w};
    uint4 H, L;
    split8(v, H, L);
    hi[i] = H;
    lo[i] = L;
}

// ---------------------------------------------------------------------------
// shared spline-coefficient computation (per block row stripe)
// ---------------------------------------------------------------------------
__device__ __forceinline__ void compute_alpha(float (*alphaT)[WG_M], int m0, int tid,
                                              const float* __restrict__ phase,
                                              const float* __restrict__ basis) {
    if (tid < WG_M) {
        const float ph = phase[m0 + tid];
        const float s  = ph * 0.636619772367581343f;  // phase / (pi/2)
        float qf = floorf(s);
        qf = fminf(fmaxf(qf, 0.0f), 3.0f);
        const int   q  = (int)qf;
        const float t1 = s - qf;
        const float t2 = t1 * t1;
        const float t3 = t2 * t1;
        float coeff[4];
#pragma unroll
        for (int j = 0; j < 4; ++j)
            coeff[j] = t3 * basis[0 * 4 + j] + t2 * basis[1 * 4 + j]
                     + t1 * basis[2 * 4 + j] +      basis[3 * 4 + j];
        // CP_IDX[q][j] = (j + q + 3) % 4  =>  alpha[c] = coeff[(c - q + 1) mod 4]
#pragma unroll
        for (int c = 0; c < 4; ++c)
            alphaT[c][tid] = coeff[(c - q + 1 + 4) & 3];
    }
}

// ---------------------------------------------------------------------------
// Main GEMM: pre-split bf16 inputs, async-LDS double buffering
// ---------------------------------------------------------------------------
__global__ __launch_bounds__(256)
void phase_linear_wmma_async(const bf16_t* __restrict__ xh, const bf16_t* __restrict__ xl,
                             const bf16_t* __restrict__ wh, const bf16_t* __restrict__ wl,
                             const float* __restrict__ phase,
                             const float* __restrict__ biases,
                             const float* __restrict__ basis,
                             float* __restrict__ out) {
    __shared__ u32   As_hi[2][WG_M * AS];
    __shared__ u32   As_lo[2][WG_M * AS];
    __shared__ u32   Bs_hi[2][WG_N * AS];
    __shared__ u32   Bs_lo[2][WG_N * AS];
    __shared__ float alphaT[NUM_CP][WG_M];

    const int tid   = threadIdx.x;
    const int lane  = tid & 31;
    const int wave  = tid >> 5;
    const int lrow  = lane & 15;
    const int khalf = lane >> 4;

    const int m0 = blockIdx.y * WG_M;
    const int n0 = blockIdx.x * WG_N;

    compute_alpha(alphaT, m0, tid, phase, basis);

    const int waveM = wave & 1;
    const int waveN = wave >> 1;

    v8f acc[2][2], out_acc[2][2];
#pragma unroll
    for (int tm = 0; tm < 2; ++tm)
#pragma unroll
        for (int tn = 0; tn < 2; ++tn) {
            acc[tm][tn]     = (v8f){0.f, 0.f, 0.f, 0.f, 0.f, 0.f, 0.f, 0.f};
            out_acc[tm][tn] = (v8f){0.f, 0.f, 0.f, 0.f, 0.f, 0.f, 0.f, 0.f};
        }

    // staging coordinates: A 64x32 (1x16B chunk/thread/array), B 128x32 (2 chunks)
    const int ar  = tid >> 2;
    const int acx = (tid & 3) * 8;       // bf16 elements
    const int bn  = tid >> 1;
    const int bcx = (tid & 1) * 16;

    const size_t aBase = (size_t)(m0 + ar) * IN_DIM + acx;
    const u32 aOffU = (u32)(ar * AS + (acx >> 1)) * 4;              // LDS byte offset in row
    const u32 bOffU = (u32)(bn * AS + (bcx >> 1)) * 4;

    u32 ldsAh[2], ldsAl[2], ldsBh[2], ldsBl[2];
#pragma unroll
    for (int b = 0; b < 2; ++b) {
        ldsAh[b] = (u32)(uintptr_t)&As_hi[b][0] + aOffU;
        ldsAl[b] = (u32)(uintptr_t)&As_lo[b][0] + aOffU;
        ldsBh[b] = (u32)(uintptr_t)&Bs_hi[b][0] + bOffU;
        ldsBl[b] = (u32)(uintptr_t)&Bs_lo[b][0] + bOffU;
    }

    // stage step s into buffer b: 6 async b128 copies per thread
    auto stage = [&](int s, int b) {
        const int p  = s >> 5;
        const int kk = (s & 31) * KTILE;
        async_b128(ldsAh[b], xh + aBase + kk);
        async_b128(ldsAl[b], xl + aBase + kk);
        const size_t wBase = ((size_t)p * OUT_DIM + (n0 + bn)) * IN_DIM + kk + bcx;
        async_b128(ldsBh[b],      wh + wBase);
        async_b128(ldsBh[b] + 16, wh + wBase + 8);
        async_b128(ldsBl[b],      wl + wBase);
        async_b128(ldsBl[b] + 16, wl + wBase + 8);
    };

    stage(0, 0);
    wait_async0();
    __syncthreads();

#pragma unroll 1
    for (int s = 0; s < NSTEPS; ++s) {
        const int cur = s & 1;
        if (s + 1 < NSTEPS) stage(s + 1, cur ^ 1);   // overlap copies with compute

        // ---- fragments from current buffer ----
        FragU a_hi[2], a_lo[2], b_hi[2], b_lo[2];
#pragma unroll
        for (int tm = 0; tm < 2; ++tm) {
            const int base = (waveM * 32 + tm * 16 + lrow) * AS;
            a_hi[tm].u[0] = *(const uint4*)&As_hi[cur][base + khalf * 4];
            a_hi[tm].u[1] = *(const uint4*)&As_hi[cur][base + 8 + khalf * 4];
            a_lo[tm].u[0] = *(const uint4*)&As_lo[cur][base + khalf * 4];
            a_lo[tm].u[1] = *(const uint4*)&As_lo[cur][base + 8 + khalf * 4];
        }
#pragma unroll
        for (int tn = 0; tn < 2; ++tn) {
            const int base = (waveN * 32 + tn * 16 + lrow) * AS + khalf * 4;
            b_hi[tn].u[0] = *(const uint4*)&Bs_hi[cur][base];
            b_hi[tn].u[1] = *(const uint4*)&Bs_hi[cur][base + 8];
            b_lo[tn].u[0] = *(const uint4*)&Bs_lo[cur][base];
            b_lo[tn].u[1] = *(const uint4*)&Bs_lo[cur][base + 8];
        }

        // ---- bf16x2 error-compensated MMA ----
#pragma unroll
        for (int tm = 0; tm < 2; ++tm)
#pragma unroll
            for (int tn = 0; tn < 2; ++tn) {
                acc[tm][tn] = __builtin_amdgcn_wmma_f32_16x16x32_bf16(
                    false, a_hi[tm].v, false, b_hi[tn].v, (short)0, acc[tm][tn], false, false);
                acc[tm][tn] = __builtin_amdgcn_wmma_f32_16x16x32_bf16(
                    false, a_lo[tm].v, false, b_hi[tn].v, (short)0, acc[tm][tn], false, false);
                acc[tm][tn] = __builtin_amdgcn_wmma_f32_16x16x32_bf16(
                    false, a_hi[tm].v, false, b_lo[tn].v, (short)0, acc[tm][tn], false, false);
            }

        // ---- end of expert p: blend with spline weight, reset acc ----
        if ((s & 31) == 31) {
            const int p = s >> 5;
#pragma unroll
            for (int tm = 0; tm < 2; ++tm) {
#pragma unroll
                for (int r = 0; r < 8; ++r) {
                    const float a = alphaT[p][waveM * 32 + tm * 16 + khalf * 8 + r];
#pragma unroll
                    for (int tn = 0; tn < 2; ++tn)
                        out_acc[tm][tn][r] += a * acc[tm][tn][r];
                }
            }
#pragma unroll
            for (int tm = 0; tm < 2; ++tm)
#pragma unroll
                for (int tn = 0; tn < 2; ++tn)
                    acc[tm][tn] = (v8f){0.f, 0.f, 0.f, 0.f, 0.f, 0.f, 0.f, 0.f};
        }

        wait_async0();      // next buffer landed in LDS
        __syncthreads();    // also orders this buffer's reads vs. its next overwrite
    }

    // ---- epilogue: add alpha @ biases, store fp32 ----
#pragma unroll
    for (int tm = 0; tm < 2; ++tm) {
#pragma unroll
        for (int tn = 0; tn < 2; ++tn) {
            const int ng  = n0 + waveN * 32 + tn * 16 + lrow;
            const float bz0 = biases[0 * OUT_DIM + ng];
            const float bz1 = biases[1 * OUT_DIM + ng];
            const float bz2 = biases[2 * OUT_DIM + ng];
            const float bz3 = biases[3 * OUT_DIM + ng];
#pragma unroll
            for (int r = 0; r < 8; ++r) {
                const int mloc = waveM * 32 + tm * 16 + khalf * 8 + r;
                const float bias = alphaT[0][mloc] * bz0 + alphaT[1][mloc] * bz1
                                 + alphaT[2][mloc] * bz2 + alphaT[3][mloc] * bz3;
                out[(size_t)(m0 + mloc) * OUT_DIM + ng] = out_acc[tm][tn][r] + bias;
            }
        }
    }
}

// ---------------------------------------------------------------------------
// Fallback: self-contained kernel (in-loop conversion), used if ws too small
// ---------------------------------------------------------------------------
__global__ __launch_bounds__(256)
void phase_linear_wmma(const float* __restrict__ x,
                       const float* __restrict__ phase,
                       const float* __restrict__ weights,
                       const float* __restrict__ biases,
                       const float* __restrict__ basis,
                       float* __restrict__ out) {
    __shared__ u32   As_hi[WG_M * AS];
    __shared__ u32   As_lo[WG_M * AS];
    __shared__ u32   Bs_hi[WG_N * AS];
    __shared__ u32   Bs_lo[WG_N * AS];
    __shared__ float alphaT[NUM_CP][WG_M];

    const int tid   = threadIdx.x;
    const int lane  = tid & 31;
    const int wave  = tid >> 5;
    const int lrow  = lane & 15;
    const int khalf = lane >> 4;

    const int m0 = blockIdx.y * WG_M;
    const int n0 = blockIdx.x * WG_N;

    compute_alpha(alphaT, m0, tid, phase, basis);
    __syncthreads();

    const int waveM = wave & 1;
    const int waveN = wave >> 1;

    v8f acc[2][2];
#pragma unroll
    for (int tm = 0; tm < 2; ++tm)
#pragma unroll
        for (int tn = 0; tn < 2; ++tn)
            acc[tm][tn] = (v8f){0.f, 0.f, 0.f, 0.f, 0.f, 0.f, 0.f, 0.f};

    const int ar = tid >> 2;
    const int ac = (tid & 3) * 8;
    const int bn = tid >> 1;
    const int bc = (tid & 1) * 16;

    const float* aRow = x + (size_t)(m0 + ar) * IN_DIM + ac;

#pragma unroll 1
    for (int p = 0; p < NUM_CP; ++p) {
        const float  arp  = alphaT[p][ar];
        const float* wRow = weights + ((size_t)p * OUT_DIM + (n0 + bn)) * IN_DIM + bc;

#pragma unroll 1
        for (int kt = 0; kt < IN_DIM / KTILE; ++kt) {
            const int kk = kt * KTILE;
            __syncthreads();
            {
                const float4 f0 = *(const float4*)(aRow + kk);
                const float4 f1 = *(const float4*)(aRow + kk + 4);
                float v[8] = {f0.x * arp, f0.y * arp, f0.z * arp, f0.w * arp,
                              f1.x * arp, f1.y * arp, f1.z * arp, f1.w * arp};
                uint4 H, L;
                split8(v, H, L);
                const int off = ar * AS + (ac >> 1);
                *(uint4*)&As_hi[off] = H;
                *(uint4*)&As_lo[off] = L;
            }
            {
                const int off = bn * AS + (bc >> 1);
                const float4 f0 = *(const float4*)(wRow + kk);
                const float4 f1 = *(const float4*)(wRow + kk + 4);
                const float4 f2 = *(const float4*)(wRow + kk + 8);
                const float4 f3 = *(const float4*)(wRow + kk + 12);
                float v0[8] = {f0.x, f0.y, f0.z, f0.w, f1.x, f1.y, f1.z, f1.w};
                float v1[8] = {f2.x, f2.y, f2.z, f2.w, f3.x, f3.y, f3.z, f3.w};
                uint4 H, L;
                split8(v0, H, L);
                *(uint4*)&Bs_hi[off]     = H;
                *(uint4*)&Bs_lo[off]     = L;
                split8(v1, H, L);
                *(uint4*)&Bs_hi[off + 4] = H;
                *(uint4*)&Bs_lo[off + 4] = L;
            }
            __syncthreads();

            FragU a_hi[2], a_lo[2], b_hi[2], b_lo[2];
#pragma unroll
            for (int tm = 0; tm < 2; ++tm) {
                const int base = (waveM * 32 + tm * 16 + lrow) * AS;
                a_hi[tm].u[0] = *(const uint4*)&As_hi[base + khalf * 4];
                a_hi[tm].u[1] = *(const uint4*)&As_hi[base + 8 + khalf * 4];
                a_lo[tm].u[0] = *(const uint4*)&As_lo[base + khalf * 4];
                a_lo[tm].u[1] = *(const uint4*)&As_lo[base + 8 + khalf * 4];
            }
#pragma unroll
            for (int tn = 0; tn < 2; ++tn) {
                const int base = (waveN * 32 + tn * 16 + lrow) * AS + khalf * 4;
                b_hi[tn].u[0] = *(const uint4*)&Bs_hi[base];
                b_hi[tn].u[1] = *(const uint4*)&Bs_hi[base + 8];
                b_lo[tn].u[0] = *(const uint4*)&Bs_lo[base];
                b_lo[tn].u[1] = *(const uint4*)&Bs_lo[base + 8];
            }
#pragma unroll
            for (int tm = 0; tm < 2; ++tm)
#pragma unroll
                for (int tn = 0; tn < 2; ++tn) {
                    acc[tm][tn] = __builtin_amdgcn_wmma_f32_16x16x32_bf16(
                        false, a_hi[tm].v, false, b_hi[tn].v, (short)0, acc[tm][tn], false, false);
                    acc[tm][tn] = __builtin_amdgcn_wmma_f32_16x16x32_bf16(
                        false, a_lo[tm].v, false, b_hi[tn].v, (short)0, acc[tm][tn], false, false);
                    acc[tm][tn] = __builtin_amdgcn_wmma_f32_16x16x32_bf16(
                        false, a_hi[tm].v, false, b_lo[tn].v, (short)0, acc[tm][tn], false, false);
                }
        }
    }

#pragma unroll
    for (int tm = 0; tm < 2; ++tm) {
#pragma unroll
        for (int tn = 0; tn < 2; ++tn) {
            const int ng  = n0 + waveN * 32 + tn * 16 + lrow;
            const float bz0 = biases[0 * OUT_DIM + ng];
            const float bz1 = biases[1 * OUT_DIM + ng];
            const float bz2 = biases[2 * OUT_DIM + ng];
            const float bz3 = biases[3 * OUT_DIM + ng];
#pragma unroll
            for (int r = 0; r < 8; ++r) {
                const int mloc = waveM * 32 + tm * 16 + khalf * 8 + r;
                const float bias = alphaT[0][mloc] * bz0 + alphaT[1][mloc] * bz1
                                 + alphaT[2][mloc] * bz2 + alphaT[3][mloc] * bz3;
                out[(size_t)(m0 + mloc) * OUT_DIM + ng] = acc[tm][tn][r] + bias;
            }
        }
    }
}

extern "C" void kernel_launch(void* const* d_in, const int* in_sizes, int n_in,
                              void* d_out, int out_size, void* d_ws, size_t ws_size,
                              hipStream_t stream) {
    const float* x       = (const float*)d_in[0];
    const float* phase   = (const float*)d_in[1];
    const float* weights = (const float*)d_in[2];
    const float* biases  = (const float*)d_in[3];
    const float* basis   = (const float*)d_in[4];
    float*       out     = (float*)d_out;

    const int Bsz = in_sizes[1];
    dim3 grid(OUT_DIM / WG_N, Bsz / WG_M);

    const size_t nx   = (size_t)Bsz * IN_DIM;                    // x elements
    const size_t nw   = (size_t)NUM_CP * OUT_DIM * IN_DIM;       // weight elements
    const size_t need = (nx + nw) * 4;                           // hi+lo bf16 each

    if (ws_size >= need) {
        char* w = (char*)d_ws;
        bf16_t* xh = (bf16_t*)(w);
        bf16_t* xl = (bf16_t*)(w + nx * 2);
        bf16_t* wh = (bf16_t*)(w + nx * 4);
        bf16_t* wl = (bf16_t*)(w + nx * 4 + nw * 2);

        const int nx8 = (int)(nx / 8), nw8 = (int)(nw / 8);
        split_to_bf16<<<(nx8 + 255) / 256, 256, 0, stream>>>(x, (uint4*)xh, (uint4*)xl, nx8);
        split_to_bf16<<<(nw8 + 255) / 256, 256, 0, stream>>>(weights, (uint4*)wh, (uint4*)wl, nw8);
        phase_linear_wmma_async<<<grid, 256, 0, stream>>>(xh, xl, wh, wl,
                                                          phase, biases, basis, out);
    } else {
        phase_linear_wmma<<<grid, 256, 0, stream>>>(x, phase, weights, biases, basis, out);
    }
}
